// Attention_78297253806423
// MI455X (gfx1250) — matrix operand order repaired and checked by reading, not compiled, not run on hardware
//
#include <hip/hip_runtime.h>

// ---------------------------------------------------------------------------
// Problem constants (from the reference)
// ---------------------------------------------------------------------------
static constexpr int    BB    = 8192;
static constexpr int    NTOK  = 17;
static constexpr int    CC    = 512;
static constexpr int    HH    = 8;
static constexpr int    HDD   = 64;
static constexpr int    MROWS = BB * NTOK;          // 139264 (divisible by 128)

static constexpr size_t XBF_ELEMS   = (size_t)MROWS * CC;          // 71,303,168
static constexpr size_t WQKV_ELEMS  = (size_t)CC * 3 * CC;         // 786,432
static constexpr size_t WPROJ_ELEMS = (size_t)CC * CC;             // 262,144
static constexpr size_t Q_ELEMS     = (size_t)BB * HH * NTOK * HDD;// 71,303,168
static constexpr size_t KT_ELEMS    = (size_t)BB * HH * HDD * 32;  // 134,217,728
static constexpr size_t V_ELEMS     = Q_ELEMS;
static constexpr size_t SLACK       = 1024;

// ---------------------------------------------------------------------------
// Feature probes (compile-safe: fallback paths keep the build green)
// ---------------------------------------------------------------------------
#if defined(__gfx1250__) && __has_builtin(__builtin_amdgcn_global_load_async_to_lds_b128) && \
    __has_builtin(__builtin_amdgcn_s_wait_asynccnt)
#define HAVE_ASYNC_LDS 1
#else
#define HAVE_ASYNC_LDS 0
#endif

#if HAVE_ASYNC_LDS && __has_builtin(__builtin_amdgcn_tensor_load_to_lds) && \
    __has_builtin(__builtin_amdgcn_s_wait_tensorcnt)
#define HAVE_TDM 1
#else
#define HAVE_TDM 0
#endif

// ---------------------------------------------------------------------------
// Vector types (gfx1250, wave32)
// ---------------------------------------------------------------------------
typedef __attribute__((ext_vector_type(16))) __bf16 v16bf;
typedef __attribute__((ext_vector_type(8)))  __bf16 v8bf;
typedef __attribute__((ext_vector_type(8)))  float  v8f;
typedef __attribute__((ext_vector_type(4)))  unsigned int v4ui;
typedef __attribute__((ext_vector_type(8)))  int    v8i_t;
typedef __attribute__((ext_vector_type(4)))  int    v4i_t;

#if HAVE_ASYNC_LDS
typedef __attribute__((address_space(1))) v4i_t* gv4i_ptr;   // global v4i*
typedef __attribute__((address_space(3))) v4i_t* lv4i_ptr;   // LDS v4i*
#endif

#define DI __device__ __forceinline__

DI unsigned short f32_to_bf16(float f) {
    unsigned int u = __float_as_uint(f);
    unsigned int r = u + 0x7FFFu + ((u >> 16) & 1u);   // round-to-nearest-even
    return (unsigned short)(r >> 16);
}

DI v16bf zero_frag() {
    union { v16bf v; unsigned int u[8]; } z;
#pragma unroll
    for (int i = 0; i < 8; ++i) z.u[i] = 0u;
    return z.v;
}

// A-matrix 16x32 bf16 fragment (ISA 7.12.2):
//   lane L (L<16): row M=L, VGPR0-3 = K 0..7,  VGPR4-7 = K 16..23
//   lane L+16    : row M=L, VGPR0-3 = K 8..15, VGPR4-7 = K 24..31
DI v16bf load_frag_a(const unsigned short* A, size_t lda, int m0, int k0, int lane) {
    const int row = m0 + (lane & 15);
    const int hi  = lane >> 4;
    const unsigned short* p = A + (size_t)row * lda + (size_t)(k0 + 8 * hi);
    union { v16bf v; v8bf h[2]; } u;
    u.h[0] = *(const v8bf*)(p);
    u.h[1] = *(const v8bf*)(p + 16);
    return u.v;
}

// B-matrix 32x16 bf16 fragment: lane L holds column N=L%16, K in same order as A
// (s<8: K=8*hi+s, s>=8: K=16+8*hi+s-8); K >= kmax is zero-filled.
DI v16bf load_frag_b(const unsigned short* Bm, size_t ldb, int k0, int n0, int lane, int kmax = 32) {
    const int col = n0 + (lane & 15);
    const int hi  = lane >> 4;
    union { v16bf v; unsigned short s[16]; } u;
#pragma unroll
    for (int s = 0; s < 16; ++s) {
        const int k = (s < 8) ? (8 * hi + s) : (16 + 8 * hi + (s - 8));
        u.s[s] = (k < kmax) ? Bm[(size_t)(k0 + k) * ldb + (size_t)col] : (unsigned short)0;
    }
    return u.v;
}

DI v8f wmma_bf16(v16bf a, v16bf b, v8f c) {
    return __builtin_amdgcn_wmma_f32_16x16x32_bf16(false, a, false, b,
                                                   (short)0, c, false, false);
}

#if HAVE_ASYNC_LDS
// Per-lane async 16-byte copy global -> LDS (ASYNCcnt tracked).
DI void async_copy16(unsigned short* lds_dst, const unsigned short* gsrc) {
    __builtin_amdgcn_global_load_async_to_lds_b128(
        (gv4i_ptr)gsrc, (lv4i_ptr)lds_dst, 0, 0);
}
#endif

#if HAVE_TDM
// TDM: load a 2D tile (tile_rows x tile_cols, bf16) from a row-major matrix
// with row stride `ld` into LDS (contiguous row-major tile). ISA 8.3-8.6.
// Toolchain builtin arity: 6 args (g0 u32x4, g1 i32x8, g2 i32x4, g3 i32x4,
// extra i32x8 zero-filled, cpol).
DI void tdm_load_tile_2d(unsigned short* lds_dst, const unsigned short* gsrc,
                         unsigned ld, unsigned tile_rows, unsigned tile_cols) {
    const unsigned long long ga = (unsigned long long)(uintptr_t)gsrc;
    const unsigned lds_off = (unsigned)((uintptr_t)lds_dst & 0xFFFFFFFFu);
    const unsigned td0 = ld;             // tensor dim0 (elements per row)
    const unsigned td1 = 1u << 20;       // generous dim1 bound (tiles stay in-bounds)
    const unsigned long long st0 = ld;   // dim0 stride (elements)

    v4ui g0;
    g0[0] = 1u;                                                // count=1, user descriptor
    g0[1] = lds_off;                                           // lds_addr (bytes)
    g0[2] = (unsigned)(ga & 0xFFFFFFFFu);                      // global_addr[31:0]
    g0[3] = (unsigned)((ga >> 32) & 0x01FFFFFFu) | (2u << 30); // addr[56:32] | type=2

    v8i_t g1;
    g1[0] = 0x00010000;                                        // data_size=2B
    g1[1] = (int)((td0 & 0xFFFFu) << 16);                      // tensor_dim0[15:0]
    g1[2] = (int)((td0 >> 16) | ((td1 & 0xFFFFu) << 16));      // dim0[31:16] | dim1[15:0]
    g1[3] = (int)((td1 >> 16) | (tile_cols << 16));            // dim1[31:16] | tile_dim0
    g1[4] = (int)tile_rows;                                    // tile_dim1 (tile_dim2=0)
    g1[5] = (int)(unsigned)(st0 & 0xFFFFFFFFu);                // dim0_stride[31:0]
    g1[6] = (int)(unsigned)((st0 >> 32) & 0xFFFFu);            // dim0_stride[47:32]
    g1[7] = 0;

    v4i_t gz4 = {0, 0, 0, 0};
    v8i_t gz8 = {0, 0, 0, 0, 0, 0, 0, 0};
    __builtin_amdgcn_tensor_load_to_lds(g0, g1, gz4, gz4, gz8, 0);
}
#endif

// ---------------------------------------------------------------------------
// Shared GEMM mainloop: 128x128 block tile, 8 waves (4x2), wave tile 32x64.
// LDB = column count of B. Accumulates into acc[2][4] (fp32).
// ---------------------------------------------------------------------------
template <int LDB>
DI void gemm128_mainloop(const unsigned short* __restrict__ A,
                         const unsigned short* __restrict__ Bm,
                         int mbase, int nbase, v8f acc[2][4])
{
    const int tid  = threadIdx.x;
    const int lane = tid & 31;
    const int wave = tid >> 5;
    const int wm = wave >> 1;   // 0..3
    const int wn = wave & 1;    // 0..1

#if HAVE_ASYNC_LDS
    __shared__ __align__(16) unsigned short sA[2 * 128 * 32];   // 16 KB
    __shared__ __align__(16) unsigned short sB[2 * 32 * 128];   // 16 KB

    auto stage = [&](int buf, int k0) {
        unsigned short* dstA = sA + buf * (128 * 32);
        unsigned short* dstB = sB + buf * (32 * 128);
        // A tile: 128 rows x 32 cols -> 512 x b128, 2 per thread
#pragma unroll
        for (int i = 0; i < 2; ++i) {
            const int idx = tid + i * 256;
            const int row = idx >> 2, seg = idx & 3;
            async_copy16(dstA + row * 32 + seg * 8,
                         A + (size_t)(mbase + row) * CC + (size_t)(k0 + seg * 8));
        }
#if HAVE_TDM
        // B tile: one TDM descriptor moves the whole 32x128 tile
        if (wave == 0)
            tdm_load_tile_2d(dstB, Bm + (size_t)k0 * LDB + nbase, LDB, 32, 128);
#else
        // B tile: 32 rows x 128 cols -> 512 x b128, 2 per thread
#pragma unroll
        for (int i = 0; i < 2; ++i) {
            const int idx = tid + i * 256;
            const int row = idx >> 4, seg = idx & 15;
            async_copy16(dstB + row * 128 + seg * 8,
                         Bm + (size_t)(k0 + row) * LDB + (size_t)(nbase + seg * 8));
        }
#endif
    };
    auto wait_stage = [&]() {
        __builtin_amdgcn_s_wait_asynccnt(0);
#if HAVE_TDM
        if (wave == 0) __builtin_amdgcn_s_wait_tensorcnt(0);
#endif
        __syncthreads();
    };

    constexpr int NK = CC / 32;
    stage(0, 0);
    wait_stage();
    for (int kk = 0; kk < NK; ++kk) {
        const int cur = kk & 1;
        if (kk + 1 < NK) stage(cur ^ 1, (kk + 1) * 32);

        const unsigned short* tA = sA + cur * (128 * 32);
        const unsigned short* tB = sB + cur * (32 * 128);
        v16bf afr[2], bfr[4];
#pragma unroll
        for (int mt = 0; mt < 2; ++mt)
            afr[mt] = load_frag_a(tA, 32, wm * 32 + mt * 16, 0, lane);
#pragma unroll
        for (int nt = 0; nt < 4; ++nt)
            bfr[nt] = load_frag_b(tB, 128, 0, wn * 64 + nt * 16, lane);
#pragma unroll
        for (int mt = 0; mt < 2; ++mt)
#pragma unroll
            for (int nt = 0; nt < 4; ++nt)
                acc[mt][nt] = wmma_bf16(afr[mt], bfr[nt], acc[mt][nt]);

        if (kk + 1 < NK) wait_stage();
    }
#else
    // Fallback: direct per-lane fragment loads from global (L0/L2 cached)
    const int m0 = mbase + wm * 32;
    const int n0 = nbase + wn * 64;
    for (int k0 = 0; k0 < CC; k0 += 32) {
        if (k0 + 32 < CC) {
            __builtin_prefetch(A + (size_t)(m0 + (lane & 15)) * CC + (k0 + 32), 0, 1);
            __builtin_prefetch(Bm + (size_t)(k0 + 32 + lane) * LDB + n0, 0, 1);
        }
        v16bf afr[2], bfr[4];
#pragma unroll
        for (int mt = 0; mt < 2; ++mt)
            afr[mt] = load_frag_a(A, CC, m0 + mt * 16, k0, lane);
#pragma unroll
        for (int nt = 0; nt < 4; ++nt)
            bfr[nt] = load_frag_b(Bm, LDB, k0, n0 + nt * 16, lane);
#pragma unroll
        for (int mt = 0; mt < 2; ++mt)
#pragma unroll
            for (int nt = 0; nt < 4; ++nt)
                acc[mt][nt] = wmma_bf16(afr[mt], bfr[nt], acc[mt][nt]);
    }
#endif
}

// ---------------------------------------------------------------------------
// Kernel 0: fp32 -> bf16 conversion (grid-stride)
// ---------------------------------------------------------------------------
__global__ void cvt_bf16_kernel(const float* __restrict__ in,
                                unsigned short* __restrict__ out, size_t n) {
    size_t i      = (size_t)blockIdx.x * blockDim.x + threadIdx.x;
    size_t stride = (size_t)gridDim.x * blockDim.x;
    for (; i < n; i += stride) out[i] = f32_to_bf16(in[i]);
}

// ---------------------------------------------------------------------------
// Kernel 1: QKV GEMM + bias + scatter to q / kT / v
// ---------------------------------------------------------------------------
__global__ __launch_bounds__(256) void qkv_wmma_kernel(
    const unsigned short* __restrict__ xbf,
    const unsigned short* __restrict__ wbf,
    const float* __restrict__ bias,
    unsigned short* __restrict__ q,
    unsigned short* __restrict__ kT,
    unsigned short* __restrict__ v)
{
    const int lane = threadIdx.x & 31;
    const int wave = threadIdx.x >> 5;
    const int m0 = blockIdx.y * 128 + (wave >> 1) * 32;
    const int n0 = blockIdx.x * 128 + (wave & 1) * 64;
    const int hi = lane >> 4;
    const int cl = lane & 15;

    v8f acc[2][4] = {};
    gemm128_mainloop<3 * CC>(xbf, wbf, blockIdx.y * 128, blockIdx.x * 128, acc);

#pragma unroll
    for (int mt = 0; mt < 2; ++mt) {
#pragma unroll
        for (int r = 0; r < 8; ++r) {
            const int gm = m0 + mt * 16 + r + 8 * hi;
            const unsigned int b  = (unsigned int)gm / 17u;
            const unsigned int tk = (unsigned int)gm % 17u;
#pragma unroll
            for (int nt = 0; nt < 4; ++nt) {
                const int gn = n0 + nt * 16 + cl;
                const unsigned short bv = f32_to_bf16(acc[mt][nt][r] + bias[gn]);
                const int which = gn >> 9;        // 0:q 1:k 2:v
                const int rem   = gn & 511;
                const int h     = rem >> 6;
                const int d     = rem & 63;
                const size_t bh = (size_t)b * HH + h;
                if (which == 0)      q [(bh * NTOK + tk) * HDD + d] = bv;
                else if (which == 1) kT[(bh * HDD + d) * 32 + tk]   = bv;
                else                 v [(bh * NTOK + tk) * HDD + d] = bv;
            }
        }
    }
}

// ---------------------------------------------------------------------------
// Kernel 2: attention. 8 waves/block, one (b,h) per wave.
// ---------------------------------------------------------------------------
__global__ __launch_bounds__(256) void attn_wmma_kernel(
    const unsigned short* __restrict__ q,
    const unsigned short* __restrict__ kT,
    const unsigned short* __restrict__ v,
    const float* __restrict__ outer,
    const float* __restrict__ alpha_p,
    unsigned short* __restrict__ attn_out)
{
    __shared__ float souter[HH * NTOK * NTOK];
    __shared__ float slog[8][32 * 32];
    __shared__ __align__(16) unsigned short sattn[8][32 * 32];

    for (int i = threadIdx.x; i < HH * NTOK * NTOK; i += 256)
        souter[i] = outer[i];
    __syncthreads();

    const int lane = threadIdx.x & 31;
    const int wave = threadIdx.x >> 5;
    const size_t idx = (size_t)blockIdx.x * 8 + wave;   // b*H + h
    const int h = (int)(idx & 7);
    const unsigned short* qb = q  + idx * (NTOK * HDD);
    const unsigned short* kb = kT + idx * (HDD * 32);
    const unsigned short* vb = v  + idx * (NTOK * HDD);
    const float alpha = alpha_p[0];
    const int hi = lane >> 4;
    const int cl = lane & 15;

    // ---- logits = q @ kT ----
    v8f accL[2][2] = {};
#pragma unroll
    for (int kk = 0; kk < 2; ++kk) {
        v16bf aq[2], bk[2];
#pragma unroll
        for (int mt = 0; mt < 2; ++mt) {
            const int row = mt * 16 + cl;
            aq[mt] = (row < NTOK) ? load_frag_a(qb, HDD, mt * 16, kk * 32, lane)
                                  : zero_frag();
        }
#pragma unroll
        for (int nt = 0; nt < 2; ++nt)
            bk[nt] = load_frag_b(kb, 32, kk * 32, nt * 16, lane);
#pragma unroll
        for (int mt = 0; mt < 2; ++mt)
#pragma unroll
            for (int nt = 0; nt < 2; ++nt)
                accL[mt][nt] = wmma_bf16(aq[mt], bk[nt], accL[mt][nt]);
    }

    // ---- (logits + alpha*outer) * scale -> LDS ----
    const float scale = 0.125f;   // 64^-0.5
#pragma unroll
    for (int mt = 0; mt < 2; ++mt)
#pragma unroll
        for (int nt = 0; nt < 2; ++nt)
#pragma unroll
            for (int r = 0; r < 8; ++r) {
                const int row = mt * 16 + r + 8 * hi;
                const int col = nt * 16 + cl;
                float val = 0.0f;
                if (row < NTOK && col < NTOK)
                    val = (accL[mt][nt][r] +
                           alpha * souter[(h * NTOK + row) * NTOK + col]) * scale;
                slog[wave][row * 32 + col] = val;
            }
    __syncthreads();

    // ---- softmax over 17 real columns; lane == row ----
    {
        const int row = lane;
        if (row < NTOK) {
            float t[NTOK];
            float mx = -3.4e38f;
#pragma unroll
            for (int c = 0; c < NTOK; ++c) {
                t[c] = slog[wave][row * 32 + c];
                mx = fmaxf(mx, t[c]);
            }
            float s = 0.0f;
#pragma unroll
            for (int c = 0; c < NTOK; ++c) { t[c] = __expf(t[c] - mx); s += t[c]; }
            const float inv = 1.0f / s;
#pragma unroll
            for (int c = 0; c < NTOK; ++c)
                sattn[wave][row * 32 + c] = f32_to_bf16(t[c] * inv);
#pragma unroll
            for (int c = NTOK; c < 32; ++c) sattn[wave][row * 32 + c] = 0;
        } else {
#pragma unroll
            for (int c = 0; c < 32; ++c) sattn[wave][row * 32 + c] = 0;
        }
    }
    __syncthreads();

    // ---- out = attn @ v ----
    v8f accO[2][4] = {};
    {
        v16bf aA[2], bV[4];
#pragma unroll
        for (int mt = 0; mt < 2; ++mt)
            aA[mt] = load_frag_a(&sattn[wave][0], 32, mt * 16, 0, lane);
#pragma unroll
        for (int nt = 0; nt < 4; ++nt)
            bV[nt] = load_frag_b(vb, HDD, 0, nt * 16, lane, NTOK);
#pragma unroll
        for (int mt = 0; mt < 2; ++mt)
#pragma unroll
            for (int nt = 0; nt < 4; ++nt)
                accO[mt][nt] = wmma_bf16(aA[mt], bV[nt], accO[mt][nt]);
    }

    const size_t b = idx >> 3;
#pragma unroll
    for (int mt = 0; mt < 2; ++mt)
#pragma unroll
        for (int r = 0; r < 8; ++r) {
            const int tok = mt * 16 + r + 8 * hi;
            if (tok < NTOK) {
#pragma unroll
                for (int nt = 0; nt < 4; ++nt) {
                    const int d = nt * 16 + cl;
                    attn_out[(b * NTOK + tok) * CC + h * HDD + d] =
                        f32_to_bf16(accO[mt][nt][r]);
                }
            }
        }
}

// ---------------------------------------------------------------------------
// Kernel 3: output projection GEMM, fp32 result to d_out
// ---------------------------------------------------------------------------
__global__ __launch_bounds__(256) void proj_wmma_kernel(
    const unsigned short* __restrict__ abf,
    const unsigned short* __restrict__ wbf,
    const float* __restrict__ bias,
    float* __restrict__ out)
{
    const int lane = threadIdx.x & 31;
    const int wave = threadIdx.x >> 5;
    const int m0 = blockIdx.y * 128 + (wave >> 1) * 32;
    const int n0 = blockIdx.x * 128 + (wave & 1) * 64;
    const int hi = lane >> 4;
    const int cl = lane & 15;

    v8f acc[2][4] = {};
    gemm128_mainloop<CC>(abf, wbf, blockIdx.y * 128, blockIdx.x * 128, acc);

#pragma unroll
    for (int mt = 0; mt < 2; ++mt)
#pragma unroll
        for (int r = 0; r < 8; ++r) {
            const size_t gm = (size_t)(m0 + mt * 16 + r + 8 * hi);
#pragma unroll
            for (int nt = 0; nt < 4; ++nt) {
                const int gn = n0 + nt * 16 + cl;
                out[gm * CC + gn] = acc[mt][nt][r] + bias[gn];
            }
        }
}

// ---------------------------------------------------------------------------
// Host launcher
// ---------------------------------------------------------------------------
extern "C" void kernel_launch(void* const* d_in, const int* in_sizes, int n_in,
                              void* d_out, int out_size, void* d_ws, size_t ws_size,
                              hipStream_t stream) {
    const float* x      = (const float*)d_in[0];
    const float* W_qkv  = (const float*)d_in[1];
    const float* b_qkv  = (const float*)d_in[2];
    const float* outer  = (const float*)d_in[3];
    const float* alpha  = (const float*)d_in[4];
    const float* W_proj = (const float*)d_in[5];
    const float* b_proj = (const float*)d_in[6];
    float* out = (float*)d_out;

    unsigned short* ws = (unsigned short*)d_ws;
    const size_t o_xbf   = 0;                                  // reused as attn_out
    const size_t o_wqkv  = o_xbf   + XBF_ELEMS;
    const size_t o_wproj = o_wqkv  + WQKV_ELEMS;
    const size_t o_q     = o_wproj + WPROJ_ELEMS;
    const size_t o_kt    = o_q     + Q_ELEMS + SLACK;
    const size_t o_v     = o_kt    + KT_ELEMS;

    cvt_bf16_kernel<<<4096, 256, 0, stream>>>(x,      ws + o_xbf,   XBF_ELEMS);
    cvt_bf16_kernel<<<512,  256, 0, stream>>>(W_qkv,  ws + o_wqkv,  WQKV_ELEMS);
    cvt_bf16_kernel<<<256,  256, 0, stream>>>(W_proj, ws + o_wproj, WPROJ_ELEMS);

    qkv_wmma_kernel<<<dim3(3 * CC / 128, MROWS / 128), 256, 0, stream>>>(
        ws + o_xbf, ws + o_wqkv, b_qkv, ws + o_q, ws + o_kt, ws + o_v);

    attn_wmma_kernel<<<(BB * HH) / 8, 256, 0, stream>>>(
        ws + o_q, ws + o_kt, ws + o_v, outer, alpha, ws + o_xbf);

    proj_wmma_kernel<<<dim3(CC / 128, MROWS / 128), 256, 0, stream>>>(
        ws + o_xbf, ws + o_wproj, b_proj, out);
}
